// TopKSAE_37580963840386
// MI455X (gfx1250) — compile-verified, matrix operand order checked by reading, not run on hardware
//
#include <hip/hip_runtime.h>

#define D_MODEL  1024
#define D_DICT   16384
#define N_TOKENS 8192
#define K_TOP    32

typedef __attribute__((ext_vector_type(16))) __bf16 v16bf;
typedef __attribute__((ext_vector_type(8)))  __bf16 v8bf;
typedef __attribute__((ext_vector_type(8)))  float  v8f;

static __device__ __forceinline__ __bf16 to_bf16(float f) {
  // round-to-nearest-even f32 -> bf16
  unsigned u = __float_as_uint(f);
  unsigned r = (u + 0x7fffu + ((u >> 16) & 1u)) >> 16;
  unsigned short s = (unsigned short)r;
  return __builtin_bit_cast(__bf16, s);
}

// low 32 bits of a generic pointer to LDS == wave-relative LDS byte address (ISA 10.2)
static __device__ __forceinline__ unsigned lds_off(const void* p) {
  return (unsigned)(unsigned long long)(uintptr_t)p;
}

#define BM 128
#define BN 128
#define BK 32
#define LDT 40                  // padded LDS row stride (bf16): 80B rows, conflict-free frags
#define TSZ (BM * LDT)          // bf16 elems per tile buffer (5120 -> 10240 B)
#define NKSTEP (D_MODEL / BK)   // 32

// one async b128: lane-private LDS byte offset + 64-bit global address
#define ASYNC_B128(ldsoff, gaddr)                                              \
  asm volatile("global_load_async_to_lds_b128 %0, %1, off"                     \
               :: "v"(ldsoff), "v"(gaddr) : "memory")

// ---------------------------------------------------------------------------
// Kernel 0: f32 -> bf16 pre-conversion (x and W_enc into workspace)
// ---------------------------------------------------------------------------
__global__ __launch_bounds__(256) void cvt_bf16_kernel(
    const float* __restrict__ s, __bf16* __restrict__ d, int n8)
{
  const int i = blockIdx.x * 256 + threadIdx.x;
  if (i >= n8) return;
  const float4* sp = (const float4*)s + (size_t)i * 2;
  const float4 a = sp[0];
  const float4 b = sp[1];
  v8bf o;
  o[0] = to_bf16(a.x); o[1] = to_bf16(a.y); o[2] = to_bf16(a.z); o[3] = to_bf16(a.w);
  o[4] = to_bf16(b.x); o[5] = to_bf16(b.y); o[6] = to_bf16(b.z); o[7] = to_bf16(b.w);
  *((v8bf*)d + i) = o;
}

// ---------------------------------------------------------------------------
// Kernel 1: z = x @ W_enc^T + b_enc  (bf16 WMMA, f32 acc, async-to-LDS
// double-buffered pipeline, one barrier per K-step)
// grid = (D_DICT/BN, N_TOKENS/BM), block = 256 (8 wave32: 4 in M x 2 in N)
// ---------------------------------------------------------------------------
__global__ __launch_bounds__(256) void sae_encode_wmma(
    const __bf16* __restrict__ xb, const __bf16* __restrict__ wb,
    const float* __restrict__ benc, float* __restrict__ z)
{
  __shared__ __bf16 As[2 * TSZ];
  __shared__ __bf16 Bs[2 * TSZ];

  const int tid  = threadIdx.x;
  const int lane = tid & 31;
  const int wave = tid >> 5;
  const int wm   = wave & 3;   // wave row: 32 M-rows each
  const int wn   = wave >> 2;  // wave col: 64 N-cols each
  const int m0   = blockIdx.y * BM;
  const int n0   = blockIdx.x * BN;

  const int lrow  = lane & 15;
  const int lhalf = lane >> 4;  // lanes 0-15: K 0..15, lanes 16-31: K 16..31

  // per-thread staging assignment: 512 b128 chunks per operand tile, 2 each.
  // chunk c: row = c>>2, col = (c&3)*8 bf16 (16 B)
  const int cA0 = tid, cA1 = tid + 256;
  const int rA0 = cA0 >> 2, colA0 = (cA0 & 3) * 8;
  const int rA1 = cA1 >> 2, colA1 = (cA1 & 3) * 8;

  const unsigned long long gA0 = (unsigned long long)(uintptr_t)(xb + (size_t)(m0 + rA0) * D_MODEL + colA0);
  const unsigned long long gA1 = (unsigned long long)(uintptr_t)(xb + (size_t)(m0 + rA1) * D_MODEL + colA1);
  const unsigned long long gB0 = (unsigned long long)(uintptr_t)(wb + (size_t)(n0 + rA0) * D_MODEL + colA0);
  const unsigned long long gB1 = (unsigned long long)(uintptr_t)(wb + (size_t)(n0 + rA1) * D_MODEL + colA1);

  const unsigned lA0 = lds_off(&As[rA0 * LDT + colA0]);
  const unsigned lA1 = lds_off(&As[rA1 * LDT + colA1]);
  const unsigned lB0 = lds_off(&Bs[rA0 * LDT + colA0]);
  const unsigned lB1 = lds_off(&Bs[rA1 * LDT + colA1]);

  auto issue_tile = [&](int k0, int buf) {
    const unsigned bofs = (unsigned)buf * (unsigned)(TSZ * 2);  // bytes
    const unsigned long long kofs = (unsigned long long)k0 * 2ull;
    ASYNC_B128(lA0 + bofs, gA0 + kofs);
    ASYNC_B128(lA1 + bofs, gA1 + kofs);
    ASYNC_B128(lB0 + bofs, gB0 + kofs);
    ASYNC_B128(lB1 + bofs, gB1 + kofs);
  };

  v8f acc[2][4] = {};

  issue_tile(0, 0);
  for (int ks = 0; ks < NKSTEP; ++ks) {
    const int buf = ks & 1;
    asm volatile("s_wait_asynccnt 0" ::: "memory");  // this wave's tile-ks lands
    __syncthreads();                                 // all waves' chunks visible;
                                                     // also: everyone done reading other buf
    if (ks + 1 < NKSTEP) issue_tile((ks + 1) * BK, (ks + 1) & 1);

    const __bf16* Ab = &As[buf * TSZ];
    const __bf16* Bb = &Bs[buf * TSZ];

    v16bf afr[2];
    v16bf bfr[4];
#pragma unroll
    for (int mi = 0; mi < 2; ++mi) {
      const __bf16* p = &Ab[(wm * 32 + mi * 16 + lrow) * LDT + lhalf * 16];
      v8bf lo = *(const v8bf*)p;
      v8bf hi = *(const v8bf*)(p + 8);
      afr[mi] = __builtin_shufflevector(lo, hi, 0,1,2,3,4,5,6,7,8,9,10,11,12,13,14,15);
    }
#pragma unroll
    for (int ni = 0; ni < 4; ++ni) {
      const __bf16* p = &Bb[(wn * 64 + ni * 16 + lrow) * LDT + lhalf * 16];
      v8bf lo = *(const v8bf*)p;
      v8bf hi = *(const v8bf*)(p + 8);
      bfr[ni] = __builtin_shufflevector(lo, hi, 0,1,2,3,4,5,6,7,8,9,10,11,12,13,14,15);
    }

#pragma unroll
    for (int mi = 0; mi < 2; ++mi)
#pragma unroll
      for (int ni = 0; ni < 4; ++ni)
        acc[mi][ni] = __builtin_amdgcn_wmma_f32_16x16x32_bf16(
            false, afr[mi], false, bfr[ni], (short)0, acc[mi][ni], false, false);
  }

  // epilogue: C/D layout is lane(0-15)->N, vgpr r -> M (= r + 8*lanehalf)
#pragma unroll
  for (int ni = 0; ni < 4; ++ni) {
    const int gn = n0 + wn * 64 + ni * 16 + lrow;
    const float bias = benc[gn];
#pragma unroll
    for (int mi = 0; mi < 2; ++mi) {
      const int gm = m0 + wm * 32 + mi * 16 + lhalf * 8;
      float* zp = z + (size_t)gm * D_DICT + gn;
      v8f a = acc[mi][ni];
#pragma unroll
      for (int r = 0; r < 8; ++r) zp[(size_t)r * D_DICT] = a[r] + bias;
    }
  }
}

// ---------------------------------------------------------------------------
// Kernel 2: abs-top-32 per row, in-place mask of z, emit (idx,val) pairs.
// One 256-thread block per token. Dynamic LDS (66 KB < 320 KB/WGP).
// Deterministic: 32 rounds of packed (key, ~idx) argmax + invalidate.
// ---------------------------------------------------------------------------
__global__ __launch_bounds__(256) void sae_topk(
    float* __restrict__ z, int* __restrict__ pidx, float* __restrict__ pval)
{
  extern __shared__ unsigned int sh[];
  unsigned int*       keys    = sh;                                   // [D_DICT]
  unsigned long long* red     = (unsigned long long*)(sh + D_DICT);   // [256]
  unsigned int*       winners = (unsigned int*)(red + 256);           // [K_TOP]

  const int row = blockIdx.x;
  const int tid = threadIdx.x;
  float* zrow = z + (size_t)row * D_DICT;

#pragma unroll 4
  for (int j = 0; j < D_DICT / 256; ++j) {
    const int i = tid + 256 * j;
    keys[i] = __float_as_uint(zrow[i]) & 0x7fffffffu;  // |z| as monotone uint
  }
  __syncthreads();

  for (int it = 0; it < K_TOP; ++it) {
    unsigned long long best = 0ull;
    for (int j = 0; j < D_DICT / 256; ++j) {
      const int i = tid + 256 * j;
      const unsigned long long p =
          ((unsigned long long)keys[i] << 32) | (unsigned)(D_DICT - 1 - i);
      best = (p > best) ? p : best;
    }
    red[tid] = best;
    __syncthreads();
    for (int s = 128; s > 0; s >>= 1) {
      if (tid < s) {
        const unsigned long long o = red[tid + s];
        if (o > red[tid]) red[tid] = o;
      }
      __syncthreads();
    }
    if (tid == 0) {
      const unsigned idx = (unsigned)(D_DICT - 1) - (unsigned)(red[0] & 0xffffffffu);
      winners[it] = idx;
      keys[idx] = 0u;  // exclude from later rounds
    }
    __syncthreads();
  }

  if (tid < K_TOP) {
    const unsigned idx = winners[tid];
    pidx[(size_t)row * K_TOP + tid] = (int)idx;
    pval[(size_t)row * K_TOP + tid] = zrow[idx];  // winners never overwritten below
  }

  // zero everything that is not a winner (winners left untouched -> z*mask)
  for (int j = 0; j < D_DICT / 256; ++j) {
    const int i = tid + 256 * j;
    bool sel = false;
#pragma unroll
    for (int t = 0; t < K_TOP; ++t) sel |= (winners[t] == (unsigned)i);
    if (!sel) zrow[i] = 0.0f;
  }
}

// ---------------------------------------------------------------------------
// Kernel 3: sparse decode  recon = sum_k val_k * W_dec[:, idx_k] + b_dec
// One block per token; each thread owns 4 of 1024 outputs; W_dec is L2-resident.
// ---------------------------------------------------------------------------
__global__ __launch_bounds__(256) void sae_decode_sparse(
    const float* __restrict__ Wdec, const float* __restrict__ bdec,
    const int* __restrict__ pidx, const float* __restrict__ pval,
    float* __restrict__ recon)
{
  __shared__ int   sidx[K_TOP];
  __shared__ float sval[K_TOP];
  const int row = blockIdx.x;
  const int tid = threadIdx.x;
  if (tid < K_TOP) {
    sidx[tid] = pidx[(size_t)row * K_TOP + tid];
    sval[tid] = pval[(size_t)row * K_TOP + tid];
  }
  __syncthreads();

  const int m0 = tid * 4;
  float a0 = bdec[m0 + 0], a1 = bdec[m0 + 1], a2 = bdec[m0 + 2], a3 = bdec[m0 + 3];
#pragma unroll 8
  for (int t = 0; t < K_TOP; ++t) {
    const int   d = sidx[t];
    const float v = sval[t];
    const float* w = Wdec + d;  // W_dec[m][d] at m*D_DICT + d
    a0 += v * w[(size_t)(m0 + 0) * D_DICT];
    a1 += v * w[(size_t)(m0 + 1) * D_DICT];
    a2 += v * w[(size_t)(m0 + 2) * D_DICT];
    a3 += v * w[(size_t)(m0 + 3) * D_DICT];
  }
  float* r = recon + (size_t)row * D_MODEL + m0;
  r[0] = a0; r[1] = a1; r[2] = a2; r[3] = a3;
}

// ---------------------------------------------------------------------------
extern "C" void kernel_launch(void* const* d_in, const int* in_sizes, int n_in,
                              void* d_out, int out_size, void* d_ws, size_t ws_size,
                              hipStream_t stream) {
  (void)in_sizes; (void)n_in; (void)out_size; (void)ws_size;
  const float* x     = (const float*)d_in[0];
  const float* W_enc = (const float*)d_in[1];
  const float* b_enc = (const float*)d_in[2];
  const float* W_dec = (const float*)d_in[3];
  const float* b_dec = (const float*)d_in[4];

  float* recon = (float*)d_out;                                   // [8192,1024]
  float* z     = (float*)d_out + (size_t)N_TOKENS * D_MODEL;      // [8192,16384]

  // workspace: bf16 x (16 MB) | bf16 W_enc (32 MB) | top-k pairs (2 MB)
  const size_t XN = (size_t)N_TOKENS * D_MODEL;   // 8.39M elems
  const size_t WN = (size_t)D_DICT * D_MODEL;     // 16.8M elems
  __bf16* xb = (__bf16*)d_ws;
  __bf16* wb = (__bf16*)((char*)d_ws + XN * sizeof(__bf16));
  int*    pidx = (int*)((char*)d_ws + (XN + WN) * sizeof(__bf16));
  float*  pval = (float*)((char*)d_ws + (XN + WN) * sizeof(__bf16)
                          + (size_t)N_TOKENS * K_TOP * sizeof(int));

  cvt_bf16_kernel<<<(int)(XN / 8 + 255) / 256, 256, 0, stream>>>(x, xb, (int)(XN / 8));
  cvt_bf16_kernel<<<(int)(WN / 8 + 255) / 256, 256, 0, stream>>>(W_enc, wb, (int)(WN / 8));

  dim3 gridE(D_DICT / BN, N_TOKENS / BM);
  sae_encode_wmma<<<gridE, 256, 0, stream>>>(xb, wb, b_enc, z);

  const size_t shbytes = (size_t)D_DICT * 4 + 256 * 8 + K_TOP * 4;  // ~66.1 KB (<320 KB/WGP)
  (void)hipFuncSetAttribute((const void*)sae_topk,
                            hipFuncAttributeMaxDynamicSharedMemorySize, (int)shbytes);
  sae_topk<<<N_TOKENS, 256, shbytes, stream>>>(z, pidx, pval);

  sae_decode_sparse<<<N_TOKENS, 256, 0, stream>>>(W_dec, b_dec, pidx, pval, recon);
}
